// Attention_66907000537586
// MI455X (gfx1250) — matrix operand order repaired and checked by reading, not compiled, not run on hardware
//
#include <hip/hip_runtime.h>
#include <hip/hip_bf16.h>

#define BATCH 8
#define CIN   256
#define HEADS 8
#define DKH   32
#define DVH   64
#define NPOS  1024
#define IDK   256
#define IDV   512
#define MTOT  (BATCH * NPOS)   // 8192 positions total

typedef __attribute__((ext_vector_type(16))) _Float16 v16h;
typedef __attribute__((ext_vector_type(8)))  _Float16 v8h;
typedef __attribute__((ext_vector_type(8)))  float    v8f;

// ---------------- WMMA fragment loaders (ISA 7.12.2 layouts) ----------------
// A-matrix 16x32 f16, from row-major [16][ld] (row = M, col = K).
// lane r=lane%16, hi=lane/16: elems 0..7 = K hi*8..hi*8+7 ; elems 8..15 = K 16+hi*8..
__device__ __forceinline__ v16h frag_load_A(const _Float16* tile, int ld) {
  int lane = threadIdx.x & 31;
  int r  = lane & 15;
  int hi = lane >> 4;
  const _Float16* p = tile + (size_t)r * ld + hi * 8;
  v8h a = *(const v8h*)(p);
  v8h b = *(const v8h*)(p + 16);
  v16h out;
#pragma unroll
  for (int i = 0; i < 8; ++i) { out[i] = a[i]; out[i + 8] = b[i]; }
  return out;
}

// B-matrix 32x16 f16 (K x N), stored as row-major [16][ld] of B^T (row = N, col = K).
// lane n=lane%16, hi=lane/16: elems 0..7 = K hi*8..hi*8+7 ; elems 8..15 = K 16+hi*8..
__device__ __forceinline__ v16h frag_load_B(const _Float16* tile, int ld) {
  int lane = threadIdx.x & 31;
  int n  = lane & 15;
  int hi = lane >> 4;
  const _Float16* p = tile + (size_t)n * ld + hi * 8;
  v8h a = *(const v8h*)(p);
  v8h b = *(const v8h*)(p + 16);
  v16h out;
#pragma unroll
  for (int i = 0; i < 8; ++i) { out[i] = a[i]; out[i + 8] = b[i]; }
  return out;
}

__device__ __forceinline__ float gelu_exact(float x) {
  return 0.5f * x * (1.0f + erff(x * 0.70710678118654752f));
}

// ---------------- Prep kernels ----------------
// x (b, c, p) f32  ->  xT (b, p, c) f16   (LDS-tiled transpose + convert)
__global__ void k_transpose_x(const float* __restrict__ x, _Float16* __restrict__ xT) {
  __shared__ float tile[32][33];
  int b  = blockIdx.z;
  int p0 = blockIdx.x * 32;
  int c0 = blockIdx.y * 32;
  for (int i = threadIdx.y; i < 32; i += 8)
    tile[i][threadIdx.x] = x[((size_t)(b * CIN + c0 + i)) * NPOS + p0 + threadIdx.x];
  __syncthreads();
  for (int i = threadIdx.y; i < 32; i += 8)
    xT[((size_t)(b * NPOS + p0 + i)) * CIN + c0 + threadIdx.x] = (_Float16)tile[threadIdx.x][i];
}

__global__ void k_cvt16(const float* __restrict__ src, _Float16* __restrict__ dst, int n) {
  int i = blockIdx.x * 256 + threadIdx.x;
  if (i < n) dst[i] = (_Float16)src[i];
}

// Fold BatchNorm (+ optional conv bias) into per-channel scale/shift.
__global__ void k_bnprep(const float* __restrict__ g,  const float* __restrict__ be,
                         const float* __restrict__ mu, const float* __restrict__ var,
                         const float* __restrict__ bias,
                         float* __restrict__ scale, float* __restrict__ shift, int n) {
  int i = blockIdx.x * 256 + threadIdx.x;
  if (i < n) {
    float sc = g[i] * rsqrtf(var[i] + 1e-5f);
    float bi = bias ? bias[i] : 0.f;
    scale[i] = sc;
    shift[i] = be[i] - mu[i] * sc + bi * sc;
  }
}

// ---------------- GEMM: Y[m][n] = sum_k A[m][k] * W[n][k] ----------------
// One wave computes a 32x64 output tile: 2 A-frags x 4 B-frags -> 8 WMMAs per K-step.
// m = global position (b*1024+pos), n = out channel. Compile-time store modes:
// MODE 0: q/k -> dst16[b][h][pos][1<<DSHIFT]   (DSHIFT=5)
// MODE 1: v   -> dst16[b][h][d][pos]           (DSHIFT=6, transposed for attention B-frags)
// MODE 2: out -> dst32[b][n][pos]              (final f32 output)
template <int MODE, int DSHIFT>
__global__ void k_gemm(const _Float16* __restrict__ A, const _Float16* __restrict__ W,
                       const float* __restrict__ scale, const float* __restrict__ shift,
                       _Float16* __restrict__ dst16, float* __restrict__ dst32, int K) {
  int m0 = blockIdx.x * 32;
  int n0 = blockIdx.y * 64;
  const _Float16* Arow0 = A + (size_t)m0 * K;
  const _Float16* Arow1 = A + (size_t)(m0 + 16) * K;
  v8f acc[8];
#pragma unroll
  for (int i = 0; i < 8; ++i) acc[i] = (v8f){};

  for (int kk = 0; kk < K; kk += 32) {
    // speculative prefetch 4 K-tiles (128 elems) ahead (global_prefetch_b8; dropped if OOB)
    __builtin_prefetch(Arow0 + kk + 128, 0, 3);
    __builtin_prefetch(W + (size_t)n0 * K + kk + 128, 0, 3);
    v16h a0 = frag_load_A(Arow0 + kk, K);
    v16h a1 = frag_load_A(Arow1 + kk, K);
    v16h bf[4];
#pragma unroll
    for (int ni = 0; ni < 4; ++ni)
      bf[ni] = frag_load_B(W + (size_t)(n0 + ni * 16) * K + kk, K);
#pragma unroll
    for (int ni = 0; ni < 4; ++ni) {
      acc[ni]     = __builtin_amdgcn_wmma_f32_16x16x32_f16(false, a0, false, bf[ni],
                                                           (short)0, acc[ni], false, false);
      acc[4 + ni] = __builtin_amdgcn_wmma_f32_16x16x32_f16(false, a1, false, bf[ni],
                                                           (short)0, acc[4 + ni], false, false);
    }
  }

  int lane = threadIdx.x & 31;
  int nn = lane & 15, hi = lane >> 4;
  float scv[4], shv[4];
#pragma unroll
  for (int ni = 0; ni < 4; ++ni) {
    int gn = n0 + ni * 16 + nn;
    scv[ni] = scale[gn];
    shv[ni] = shift[gn];
  }
#pragma unroll
  for (int mi = 0; mi < 2; ++mi) {
#pragma unroll
    for (int ni = 0; ni < 4; ++ni) {
      int gn = n0 + ni * 16 + nn;
      int h = gn >> DSHIFT;
      int d = gn & ((1 << DSHIFT) - 1);
#pragma unroll
      for (int r = 0; r < 8; ++r) {
        int gm  = m0 + mi * 16 + r + 8 * hi;
        int b   = gm >> 10;
        int pos = gm & 1023;
        float y = acc[mi * 4 + ni][r] * scv[ni] + shv[ni];
        if (MODE == 0) {
          dst16[((((size_t)(b * HEADS + h)) * NPOS + pos) << DSHIFT) + d] = (_Float16)y;
        } else if (MODE == 1) {
          dst16[((((size_t)(b * HEADS + h)) << DSHIFT) + d) * NPOS + pos] = (_Float16)y;
        } else {
          dst32[((size_t)(b * CIN + gn)) * NPOS + pos] = y;
        }
      }
    }
  }
}

// ---------------- Attention: full score slab in LDS (no online rescaling) ----------------
// grid = B*H*(1024/32), block = 128 (4 waves). LDS: S 32x1024 f32 + P 32x1024 f16 + reductions.
#define ATTN_SMEM 197760
__global__ void k_attn(const _Float16* __restrict__ Q, const _Float16* __restrict__ Km,
                       const _Float16* __restrict__ Vt, const float* __restrict__ pos_emb,
                       const int* __restrict__ pos_idx, _Float16* __restrict__ att) {
  extern __shared__ char smem[];
  float*    S      = (float*)smem;                    // 32 x 1024 f32 (128 KB)
  _Float16* P      = (_Float16*)(smem + 131072);      // 32 x 1024 f16 (64 KB)
  float*    red    = (float*)(smem + 196608);         // 128 partial maxes
  float*    red2   = red + 128;                       // 128 partial sums
  float*    rowsum = red2 + 128;                      // 32 row sums

  int bid = blockIdx.x;
  int qt  = bid & 31;
  int h   = (bid >> 5) & 7;
  int b   = bid >> 8;

  const _Float16* qbase = Q  + (((size_t)(b * HEADS + h)) * NPOS + qt * 32) * DKH;
  const _Float16* kbase = Km + ((size_t)(b * HEADS + h)) * NPOS * DKH;
  const _Float16* vbase = Vt + ((size_t)(b * HEADS + h)) * DVH * NPOS;

  int tid  = threadIdx.x;
  int wave = tid >> 5;
  int lane = tid & 31;
  int nn = lane & 15, hi = lane >> 4;

  // ---- Phase 1: S = Q K^T (each wave covers nt = wave, wave+4, ...) ----
  v16h qf0 = frag_load_A(qbase, DKH);
  v16h qf1 = frag_load_A(qbase + 16 * DKH, DKH);
  for (int nt = wave; nt < 64; nt += 4) {
    v16h kf = frag_load_B(kbase + (size_t)nt * 16 * DKH, DKH);
    v8f c0 = {}, c1 = {};
    c0 = __builtin_amdgcn_wmma_f32_16x16x32_f16(false, qf0, false, kf, (short)0, c0, false, false);
    c1 = __builtin_amdgcn_wmma_f32_16x16x32_f16(false, qf1, false, kf, (short)0, c1, false, false);
#pragma unroll
    for (int r = 0; r < 8; ++r) {
      S[(r + 8 * hi) * NPOS + nt * 16 + nn]      = c0[r];
      S[(16 + r + 8 * hi) * NPOS + nt * 16 + nn] = c1[r];
    }
  }
  __syncthreads();

  // ---- Phase 2: softmax with relative-position bias ----
  const float scl  = 0.17677669529663687f;   // dk^-0.5
  const float iscl = 5.6568542494923806f;    // dk^0.5  (bias divided by scale in ref)
  int row  = tid >> 2;
  int qp   = tid & 3;
  int grow = qt * 32 + row;
  const int* irow  = pos_idx + (size_t)grow * NPOS;
  float*     srow  = S + row * NPOS;
  _Float16*  prow  = P + row * NPOS;
  int j0 = qp * 256;

  float m = -3.4e38f;
  for (int j = j0; j < j0 + 256; ++j) {
    float lg = srow[j] * scl + pos_emb[irow[j] * HEADS + h] * iscl;
    srow[j] = lg;
    m = fmaxf(m, lg);
  }
  red[row * 4 + qp] = m;
  __syncthreads();
  float rm = fmaxf(fmaxf(red[row * 4], red[row * 4 + 1]),
                   fmaxf(red[row * 4 + 2], red[row * 4 + 3]));
  float s = 0.f;
  for (int j = j0; j < j0 + 256; ++j) {
    float e = __expf(srow[j] - rm);
    s += e;
    prow[j] = (_Float16)e;
  }
  red2[row * 4 + qp] = s;
  __syncthreads();
  if (qp == 0)
    rowsum[row] = red2[row * 4] + red2[row * 4 + 1] + red2[row * 4 + 2] + red2[row * 4 + 3];
  __syncthreads();

  // ---- Phase 3: O = P V. Wave w owns DV column block ntv=w, both 16-row blocks,
  //      so the V B-fragment is loaded once and shared by two WMMAs per K-step. ----
  {
    int ntv = wave;                       // 4 waves x 16 cols = DVH
    v8f acc0 = {}, acc1 = {};
    const _Float16* vb = vbase + (size_t)(ntv * 16) * NPOS;
    for (int kc = 0; kc < 32; ++kc) {
      v16h bf  = frag_load_B(vb + kc * 32, NPOS);
      v16h af0 = frag_load_A(P + kc * 32, NPOS);
      v16h af1 = frag_load_A(P + 16 * NPOS + kc * 32, NPOS);
      acc0 = __builtin_amdgcn_wmma_f32_16x16x32_f16(false, af0, false, bf, (short)0, acc0,
                                                    false, false);
      acc1 = __builtin_amdgcn_wmma_f32_16x16x32_f16(false, af1, false, bf, (short)0, acc1,
                                                    false, false);
    }
#pragma unroll
    for (int r = 0; r < 8; ++r) {
      int lrow0 = r + 8 * hi;
      int lrow1 = 16 + r + 8 * hi;
      float o0 = gelu_exact(acc0[r] / rowsum[lrow0]);
      float o1 = gelu_exact(acc1[r] / rowsum[lrow1]);
      size_t gm0 = (size_t)(b * NPOS + qt * 32 + lrow0);
      size_t gm1 = (size_t)(b * NPOS + qt * 32 + lrow1);
      att[gm0 * IDV + h * DVH + ntv * 16 + nn] = (_Float16)o0;
      att[gm1 * IDV + h * DVH + ntv * 16 + nn] = (_Float16)o1;
    }
  }
}

// ---------------- Launch ----------------
extern "C" void kernel_launch(void* const* d_in, const int* in_sizes, int n_in,
                              void* d_out, int out_size, void* d_ws, size_t ws_size,
                              hipStream_t stream) {
  (void)in_sizes; (void)n_in; (void)out_size; (void)ws_size;
  const float* x    = (const float*)d_in[0];
  const float* wq   = (const float*)d_in[1];
  const float* qgam = (const float*)d_in[2];
  const float* qbet = (const float*)d_in[3];
  const float* qmu  = (const float*)d_in[4];
  const float* qvar = (const float*)d_in[5];
  const float* wk   = (const float*)d_in[6];
  const float* kgam = (const float*)d_in[7];
  const float* kbet = (const float*)d_in[8];
  const float* kmu  = (const float*)d_in[9];
  const float* kvar = (const float*)d_in[10];
  const float* wv   = (const float*)d_in[11];
  const float* vgam = (const float*)d_in[12];
  const float* vbet = (const float*)d_in[13];
  const float* vmu  = (const float*)d_in[14];
  const float* vvar = (const float*)d_in[15];
  const float* wo   = (const float*)d_in[16];
  const float* bo   = (const float*)d_in[17];
  const float* ogam = (const float*)d_in[18];
  const float* obet = (const float*)d_in[19];
  const float* omu  = (const float*)d_in[20];
  const float* ovar = (const float*)d_in[21];
  const float* pos_emb = (const float*)d_in[22];
  const int*   pos_idx = (const int*)d_in[23];

  char* ws = (char*)d_ws;
  size_t off = 0;
  auto alloc = [&](size_t bytes) {
    char* p = ws + off;
    off += (bytes + 255) & ~(size_t)255;
    return p;
  };
  _Float16* xT   = (_Float16*)alloc((size_t)BATCH * NPOS * CIN * 2);
  _Float16* wq16 = (_Float16*)alloc((size_t)IDK * CIN * 2);
  _Float16* wk16 = (_Float16*)alloc((size_t)IDK * CIN * 2);
  _Float16* wv16 = (_Float16*)alloc((size_t)IDV * CIN * 2);
  _Float16* wo16 = (_Float16*)alloc((size_t)CIN * IDV * 2);
  float* qsc = (float*)alloc(IDK * 4); float* qsh = (float*)alloc(IDK * 4);
  float* ksc = (float*)alloc(IDK * 4); float* ksh = (float*)alloc(IDK * 4);
  float* vsc = (float*)alloc(IDV * 4); float* vsh = (float*)alloc(IDV * 4);
  float* osc = (float*)alloc(CIN * 4); float* osh = (float*)alloc(CIN * 4);
  _Float16* qbuf = (_Float16*)alloc((size_t)BATCH * HEADS * NPOS * DKH * 2);
  _Float16* kbuf = (_Float16*)alloc((size_t)BATCH * HEADS * NPOS * DKH * 2);
  _Float16* vbuf = (_Float16*)alloc((size_t)BATCH * HEADS * DVH * NPOS * 2);
  _Float16* attb = (_Float16*)alloc((size_t)BATCH * NPOS * IDV * 2);

  // Stage 0: prep
  k_transpose_x<<<dim3(NPOS / 32, CIN / 32, BATCH), dim3(32, 8), 0, stream>>>(x, xT);
  k_cvt16<<<(IDK * CIN + 255) / 256, 256, 0, stream>>>(wq, wq16, IDK * CIN);
  k_cvt16<<<(IDK * CIN + 255) / 256, 256, 0, stream>>>(wk, wk16, IDK * CIN);
  k_cvt16<<<(IDV * CIN + 255) / 256, 256, 0, stream>>>(wv, wv16, IDV * CIN);
  k_cvt16<<<(CIN * IDV + 255) / 256, 256, 0, stream>>>(wo, wo16, CIN * IDV);
  k_bnprep<<<1, 256, 0, stream>>>(qgam, qbet, qmu, qvar, nullptr, qsc, qsh, IDK);
  k_bnprep<<<1, 256, 0, stream>>>(kgam, kbet, kmu, kvar, nullptr, ksc, ksh, IDK);
  k_bnprep<<<2, 256, 0, stream>>>(vgam, vbet, vmu, vvar, nullptr, vsc, vsh, IDV);
  k_bnprep<<<1, 256, 0, stream>>>(ogam, obet, omu, ovar, bo, osc, osh, CIN);

  // Stage 1: QKV projections (WMMA GEMMs, BN folded); 32x64 tile per wave
  k_gemm<0, 5><<<dim3(MTOT / 32, IDK / 64), 32, 0, stream>>>(xT, wq16, qsc, qsh, qbuf,
                                                             nullptr, CIN);
  k_gemm<0, 5><<<dim3(MTOT / 32, IDK / 64), 32, 0, stream>>>(xT, wk16, ksc, ksh, kbuf,
                                                             nullptr, CIN);
  k_gemm<1, 6><<<dim3(MTOT / 32, IDV / 64), 32, 0, stream>>>(xT, wv16, vsc, vsh, vbuf,
                                                             nullptr, CIN);

  // Stage 2: attention (score slab + softmax in 192KB of CDNA5's 320KB LDS)
  hipFuncSetAttribute((const void*)k_attn, hipFuncAttributeMaxDynamicSharedMemorySize,
                      ATTN_SMEM);
  k_attn<<<BATCH * HEADS * (NPOS / 32), 128, ATTN_SMEM, stream>>>(qbuf, kbuf, vbuf,
                                                                  pos_emb, pos_idx, attb);

  // Stage 3: output projection (bias + BN folded), f32 output
  k_gemm<2, 0><<<dim3(MTOT / 32, CIN / 64), 32, 0, stream>>>(attb, wo16, osc, osh, nullptr,
                                                             (float*)d_out, IDV);
}
